// GNNFiloPriori_51445118271858
// MI455X (gfx1250) — matrix operand order repaired and checked by reading, not compiled, not run on hardware
//
#include <hip/hip_runtime.h>

// ---------------- model constants (match reference) ----------------
#define FEAT 512
#define DH   256      // HID
#define NH   4        // HEADS
#define HC4  1024     // HEADS*HID
#define NCLS 7

typedef __attribute__((ext_vector_type(16))) _Float16 v16h;
typedef __attribute__((ext_vector_type(8)))  _Float16 v8h;
typedef __attribute__((ext_vector_type(8)))  float    v8f;

__device__ __forceinline__ float gelu_f(float x) {
  return 0.5f * x * (1.0f + erff(x * 0.70710678118654752440f));
}
__device__ __forceinline__ void atomicMaxF(float* a, float v) {
  // order-preserving float max via int/uint atomics (init must be -inf)
  if (v >= 0.0f) atomicMax((int*)a, __float_as_int(v));
  else           atomicMin((unsigned int*)a, __float_as_uint(v));
}

// ---------------- elementwise / conversion kernels ----------------
__global__ void k_f32_to_f16(const float* __restrict__ s, _Float16* __restrict__ d, int n) {
  int i = blockIdx.x * blockDim.x + threadIdx.x;
  if (i < n) d[i] = (_Float16)s[i];
}

// src: [K, ld] f32 row-major (use columns coloff..coloff+Nc) -> dst: [Nc, K] f16
__global__ void k_transpose16(const float* __restrict__ s, int ld, int coloff,
                              int K, int Nc, _Float16* __restrict__ d) {
  int i = blockIdx.x * blockDim.x + threadIdx.x;
  if (i >= K * Nc) return;
  int n = i / K, k = i - n * K;
  d[(size_t)n * K + k] = (_Float16)s[(size_t)k * ld + coloff + n];
}

__global__ void k_fill_f32(float* __restrict__ p, float v, int n) {
  int i = blockIdx.x * blockDim.x + threadIdx.x;
  if (i < n) p[i] = v;
}

// out16[i] = elu(in32[i] + bias[col])
__global__ void k_bias_elu_f16(const float* __restrict__ in32, const float* __restrict__ bias,
                               _Float16* __restrict__ out16, int rows, int cols) {
  int i = blockIdx.x * blockDim.x + threadIdx.x;
  if (i >= rows * cols) return;
  int c = i % cols;
  float x = in32[i] + bias[c];
  out16[i] = (_Float16)((x > 0.0f) ? x : (expf(x) - 1.0f));
}

// ---------------- WMMA GEMM: out = act(A[MxK] @ BT[NxK]^T + bias) ----------------
// A row-major f16; BT is B pre-transposed ([N,K] row-major, f16).
// 2x2 register blocking: one wave -> 32x32 output (4 accumulators),
// 8 waves/block (4x2) -> 128x64 block tile; K step 32.
// Per K-step: 8x b128 loads feed 4 WMMAs (2x better wmma:vmem than 1x1).
__global__ void k_gemm_wmma(const _Float16* __restrict__ A, const _Float16* __restrict__ BT,
                            const float* __restrict__ bias,
                            float* __restrict__ out32, _Float16* __restrict__ out16,
                            int M, int N, int K, int act) {
  const int wave = threadIdx.x >> 5;
  const int lane = threadIdx.x & 31;
  const int half = lane >> 4;   // lane 0-15 vs 16-31
  const int l    = lane & 15;
  const int m0 = blockIdx.x * 128 + (wave >> 1) * 32;
  const int n0 = blockIdx.y * 64  + (wave & 1) * 32;

  int ar0 = m0 + l;      if (ar0 >= M) ar0 = M - 1;   // clamp: keeps EXEC all-ones
  int ar1 = m0 + 16 + l; if (ar1 >= M) ar1 = M - 1;
  const _Float16* A0 = A  + (size_t)ar0 * K;
  const _Float16* A1 = A  + (size_t)ar1 * K;
  const _Float16* B0 = BT + (size_t)(n0 + l) * K;
  const _Float16* B1 = BT + (size_t)(n0 + 16 + l) * K;

  v8f acc00 = {}, acc01 = {}, acc10 = {}, acc11 = {};
  for (int k0 = 0; k0 < K; k0 += 32) {
    // A 16x32 f16 fragment: lanes<16 hold K 0..15, lanes>=16 K 16..31
    union { v16h v; v8h h[2]; } a0, a1;
    a0.h[0] = *(const v8h*)(A0 + k0 + half * 16);
    a0.h[1] = *(const v8h*)(A0 + k0 + half * 16 + 8);
    a1.h[0] = *(const v8h*)(A1 + k0 + half * 16);
    a1.h[1] = *(const v8h*)(A1 + k0 + half * 16 + 8);
    // B 32x16 f16 fragment: lanes<16 hold K 0..15 of its column, lanes>=16 K 16..31
    v16h b0 = *(const v16h*)(B0 + k0 + half * 16);
    v16h b1 = *(const v16h*)(B1 + k0 + half * 16);
    if (k0 + 32 < K) {
      __builtin_prefetch(A0 + k0 + 32, 0, 1);   // global_prefetch_b8
      __builtin_prefetch(A1 + k0 + 32, 0, 1);
      __builtin_prefetch(B0 + k0 + 32, 0, 1);
      __builtin_prefetch(B1 + k0 + 32, 0, 1);
    }
    acc00 = __builtin_amdgcn_wmma_f32_16x16x32_f16(false, a0.v, false, b0, (short)0, acc00, false, false);
    acc01 = __builtin_amdgcn_wmma_f32_16x16x32_f16(false, a0.v, false, b1, (short)0, acc01, false, false);
    acc10 = __builtin_amdgcn_wmma_f32_16x16x32_f16(false, a1.v, false, b0, (short)0, acc10, false, false);
    acc11 = __builtin_amdgcn_wmma_f32_16x16x32_f16(false, a1.v, false, b1, (short)0, acc11, false, false);
  }
  const float bv0 = bias ? bias[n0 + l]      : 0.0f;
  const float bv1 = bias ? bias[n0 + 16 + l] : 0.0f;
#pragma unroll
  for (int r = 0; r < 8; ++r) {
    // C/D layout: VGPR r holds M=r (lanes<16) / M=8+r (lanes>=16), N=lane&15
    int row0 = m0 + half * 8 + r;
    int row1 = row0 + 16;
    if (row0 < M) {
      float v0 = acc00[r] + bv0; if (act == 1) v0 = gelu_f(v0);
      float v1 = acc01[r] + bv1; if (act == 1) v1 = gelu_f(v1);
      size_t o = (size_t)row0 * N + n0 + l;
      if (out32) { out32[o] = v0; out32[o + 16] = v1; }
      if (out16) { out16[o] = (_Float16)v0; out16[o + 16] = (_Float16)v1; }
    }
    if (row1 < M) {
      float v0 = acc10[r] + bv0; if (act == 1) v0 = gelu_f(v0);
      float v1 = acc11[r] + bv1; if (act == 1) v1 = gelu_f(v1);
      size_t o = (size_t)row1 * N + n0 + l;
      if (out32) { out32[o] = v0; out32[o + 16] = v1; }
      if (out16) { out16[o] = (_Float16)v0; out16[o + 16] = (_Float16)v1; }
    }
  }
}

// ---------------- LayerNorm (one wave per row, D=256) ----------------
// x = a (+b) (+biasvec);  y = LN(x)*g + beta;  out = (resid? resid + y : y)
__global__ void k_ln(const float* __restrict__ a, const float* __restrict__ b,
                     const float* __restrict__ biasvec,
                     const float* __restrict__ g, const float* __restrict__ beta,
                     const float* __restrict__ resid,
                     float* __restrict__ out32, int ld32,
                     _Float16* __restrict__ out16, int ld16,
                     int rows, int D) {
  int wid  = (blockIdx.x * blockDim.x + threadIdx.x) >> 5;
  int lane = threadIdx.x & 31;
  if (wid >= rows) return;
  const int per = D / 32;     // 8
  float xv[8], sum = 0.0f, sq = 0.0f;
  for (int i = 0; i < per; ++i) {
    int c = lane + i * 32;
    float v = a[(size_t)wid * D + c];
    if (b)       v += b[(size_t)wid * D + c];
    if (biasvec) v += biasvec[c];
    xv[i] = v; sum += v; sq += v * v;
  }
  for (int off = 16; off > 0; off >>= 1) { sum += __shfl_xor(sum, off); sq += __shfl_xor(sq, off); }
  float mean = sum / D;
  float inv  = rsqrtf(sq / D - mean * mean + 1e-5f);
  for (int i = 0; i < per; ++i) {
    int c = lane + i * 32;
    float y = (xv[i] - mean) * inv * g[c] + beta[c];
    if (resid) y += resid[(size_t)wid * D + c];
    if (out32) out32[(size_t)wid * ld32 + c] = y;
    if (out16) out16[(size_t)wid * ld16 + c] = (_Float16)y;
  }
}

// ---------------- GAT helpers ----------------
// a_src[n,h] = sum_c xw16[n,h,c]*att_s[h,c] ; same for a_dst (wave per (n,h))
__global__ void k_att_reduce(const _Float16* __restrict__ xw,
                             const float* __restrict__ as, const float* __restrict__ ad,
                             float* __restrict__ a_src, float* __restrict__ a_dst,
                             int Nn, int H, int C) {
  int wid  = (blockIdx.x * blockDim.x + threadIdx.x) >> 5;
  int lane = threadIdx.x & 31;
  if (wid >= Nn * H) return;
  int n = wid / H, h = wid - n * H;
  float ss = 0.0f, sd = 0.0f;
  for (int c = lane; c < C; c += 32) {
    float v = (float)xw[(size_t)n * H * C + h * C + c];
    ss += v * as[h * C + c];
    sd += v * ad[h * C + c];
  }
  for (int off = 16; off > 0; off >>= 1) { ss += __shfl_xor(ss, off); sd += __shfl_xor(sd, off); }
  if (lane == 0) { a_src[wid] = ss; a_dst[wid] = sd; }
}

__device__ __forceinline__ int edge_src(const int* ei, int E0, int e) { return (e < E0) ? ei[e]      : (e - E0); }
__device__ __forceinline__ int edge_dst(const int* ei, int E0, int e) { return (e < E0) ? ei[E0 + e] : (e - E0); }

__global__ void k_edge_max(const int* __restrict__ ei, int E0, int Et,
                           const float* __restrict__ a_src, const float* __restrict__ a_dst,
                           float* __restrict__ m, int H) {
  int i = blockIdx.x * blockDim.x + threadIdx.x;
  if (i >= Et * H) return;
  int e = i / H, h = i - e * H;
  int s = edge_src(ei, E0, e), d = edge_dst(ei, E0, e);
  float x = a_src[s * H + h] + a_dst[d * H + h];
  x = (x > 0.0f) ? x : 0.2f * x;                 // leaky_relu 0.2
  atomicMaxF(&m[d * H + h], x);
}

__global__ void k_edge_expsum(const int* __restrict__ ei, int E0, int Et,
                              const float* __restrict__ a_src, const float* __restrict__ a_dst,
                              const float* __restrict__ m, float* __restrict__ den,
                              float* __restrict__ ee, int H) {
  int i = blockIdx.x * blockDim.x + threadIdx.x;
  if (i >= Et * H) return;
  int e = i / H, h = i - e * H;
  int s = edge_src(ei, E0, e), d = edge_dst(ei, E0, e);
  float x = a_src[s * H + h] + a_dst[d * H + h];
  x = (x > 0.0f) ? x : 0.2f * x;
  float v = expf(x - m[d * H + h]);
  ee[i] = v;
  atomicAdd(&den[d * H + h], v);
}

// one block per edge; out[dst, :] += xw[src, :] * alpha[e, h]
__global__ void k_edge_scatter(const int* __restrict__ ei, int E0,
                               const _Float16* __restrict__ xw, const float* __restrict__ ee,
                               const float* __restrict__ den, float* __restrict__ out,
                               int H, int C) {
  int e = blockIdx.x;
  int s = edge_src(ei, E0, e), d = edge_dst(ei, E0, e);
  int HCt = H * C;
  for (int j = threadIdx.x; j < HCt; j += blockDim.x) {
    int h = j / C;
    float alpha = ee[(size_t)e * H + h] / den[d * H + h];
    atomicAdd(&out[(size_t)d * HCt + j], (float)xw[(size_t)s * HCt + j] * alpha);
  }
}

// ---------------- final tiny GEMM (N=7): plain VALU ----------------
__global__ void k_final_cls(const _Float16* __restrict__ z2, const float* __restrict__ w3,
                            const float* __restrict__ b3, float* __restrict__ out, int Nn) {
  int i = blockIdx.x * blockDim.x + threadIdx.x;
  if (i >= Nn * NCLS) return;
  int n = i / NCLS, c = i - n * NCLS;
  float s = b3[c];
#pragma unroll
  for (int k = 0; k < 64; ++k) s += (float)z2[(size_t)n * 64 + k] * w3[k * NCLS + c];
  out[i] = s;
}

// ---------------- host side ----------------
extern "C" void kernel_launch(void* const* d_in, const int* in_sizes, int n_in,
                              void* d_out, int out_size, void* d_ws, size_t ws_size,
                              hipStream_t stream) {
  const float* x  = (const float*)d_in[0];
  const int*   ei = (const int*)d_in[1];
  const int NN = in_sizes[0] / FEAT;        // 20000
  const int E0 = in_sizes[1] / 2;           // 320000
  const int ET = E0 + NN;                   // + self loops

  // params (setup_inputs dict order)
  const float* ns_in_w = (const float*)d_in[2];  const float* ns_in_b = (const float*)d_in[3];
  const float* ns0_w1 = (const float*)d_in[4];   const float* ns0_b1 = (const float*)d_in[5];
  const float* ns0_w2 = (const float*)d_in[6];   const float* ns0_b2 = (const float*)d_in[7];
  const float* ns0_lng = (const float*)d_in[8];  const float* ns0_lnb = (const float*)d_in[9];
  const float* ns1_w1 = (const float*)d_in[10];  const float* ns1_b1 = (const float*)d_in[11];
  const float* ns1_w2 = (const float*)d_in[12];  const float* ns1_b2 = (const float*)d_in[13];
  const float* ns1_lng = (const float*)d_in[14]; const float* ns1_lnb = (const float*)d_in[15];
  const float* ns_out_g = (const float*)d_in[16]; const float* ns_out_b = (const float*)d_in[17];
  const float* ss_in_w = (const float*)d_in[18]; const float* ss_in_b = (const float*)d_in[19];
  const float* g1_w = (const float*)d_in[20];    const float* g1_as = (const float*)d_in[21];
  const float* g1_ad = (const float*)d_in[22];   const float* g1_bias = (const float*)d_in[23];
  const float* g2_w = (const float*)d_in[24];    const float* g2_as = (const float*)d_in[25];
  const float* g2_ad = (const float*)d_in[26];   const float* g2_bias = (const float*)d_in[27];
  const float* ss_out_g = (const float*)d_in[28]; const float* ss_out_b = (const float*)d_in[29];
  const float* m1_wqkv = (const float*)d_in[30]; const float* m1_bqkv = (const float*)d_in[31];
  const float* m1_wo = (const float*)d_in[32];   const float* m1_bo = (const float*)d_in[33];
  const float* m2_wqkv = (const float*)d_in[34]; const float* m2_bqkv = (const float*)d_in[35];
  const float* m2_wo = (const float*)d_in[36];   const float* m2_bo = (const float*)d_in[37];
  const float* fn_g = (const float*)d_in[38];    const float* fn_b = (const float*)d_in[39];
  const float* fs_g = (const float*)d_in[40];    const float* fs_b = (const float*)d_in[41];
  const float* fu_w = (const float*)d_in[42];    const float* fu_b = (const float*)d_in[43];
  const float* cl_w1 = (const float*)d_in[44];   const float* cl_b1 = (const float*)d_in[45];
  const float* cl_w2 = (const float*)d_in[46];   const float* cl_b2 = (const float*)d_in[47];
  const float* cl_w3 = (const float*)d_in[48];   const float* cl_b3 = (const float*)d_in[49];

  // ---- workspace bump allocator (256B aligned) ----
  size_t cur = 0;
  auto alloc = [&](size_t bytes) -> char* {
    char* p = (char*)d_ws + cur;
    cur += (bytes + 255) & ~(size_t)255;
    return p;
  };
  _Float16* x16   = (_Float16*)alloc((size_t)NN * FEAT * 2);
  // transposed f16 weights
  _Float16* wt_nsin = (_Float16*)alloc((size_t)DH * FEAT * 2);
  _Float16* wt_n0w1 = (_Float16*)alloc((size_t)HC4 * DH * 2);
  _Float16* wt_n0w2 = (_Float16*)alloc((size_t)DH * HC4 * 2);
  _Float16* wt_n1w1 = (_Float16*)alloc((size_t)HC4 * DH * 2);
  _Float16* wt_n1w2 = (_Float16*)alloc((size_t)DH * HC4 * 2);
  _Float16* wt_ssin = (_Float16*)alloc((size_t)DH * FEAT * 2);
  _Float16* wt_g1   = (_Float16*)alloc((size_t)HC4 * DH * 2);
  _Float16* wt_g2   = (_Float16*)alloc((size_t)DH * HC4 * 2);
  _Float16* wt_m1v  = (_Float16*)alloc((size_t)DH * DH * 2);
  _Float16* wt_m1o  = (_Float16*)alloc((size_t)DH * DH * 2);
  _Float16* wt_m2v  = (_Float16*)alloc((size_t)DH * DH * 2);
  _Float16* wt_m2o  = (_Float16*)alloc((size_t)DH * DH * 2);
  _Float16* wt_fu   = (_Float16*)alloc((size_t)512 * 512 * 2);
  _Float16* wt_cl1  = (_Float16*)alloc((size_t)128 * 512 * 2);
  _Float16* wt_cl2  = (_Float16*)alloc((size_t)64 * 128 * 2);
  // activations
  float*    h32   = (float*)alloc((size_t)NN * DH * 4);       // h
  _Float16* h16   = (_Float16*)alloc((size_t)NN * DH * 2);    // h16 / s16
  _Float16* big16 = (_Float16*)alloc((size_t)NN * HC4 * 2);   // t16 / xw1 / s2 (sequential reuse)
  float*    big32 = (float*)alloc((size_t)NN * HC4 * 4);      // gat1 out / gat2 out
  float*    tmp32 = (float*)alloc((size_t)NN * DH * 4);       // br / attn-out
  float*    node32 = (float*)alloc((size_t)NN * DH * 4);
  _Float16* node16 = (_Float16*)alloc((size_t)NN * DH * 2);
  _Float16* xw2_16 = (_Float16*)alloc((size_t)NN * DH * 2);
  float* a_s1 = (float*)alloc((size_t)NN * NH * 4);
  float* a_d1 = (float*)alloc((size_t)NN * NH * 4);
  float* mx1  = (float*)alloc((size_t)NN * NH * 4);
  float* dn1  = (float*)alloc((size_t)NN * NH * 4);
  float* ee1  = (float*)alloc((size_t)ET * NH * 4);
  float* a_s2 = (float*)alloc((size_t)NN * 4);
  float* a_d2 = (float*)alloc((size_t)NN * 4);
  float* mx2  = (float*)alloc((size_t)NN * 4);
  float* dn2  = (float*)alloc((size_t)NN * 4);
  float* ee2  = (float*)alloc((size_t)ET * 4);
  float*    st32 = (float*)alloc((size_t)NN * DH * 4);
  _Float16* st16 = (_Float16*)alloc((size_t)NN * DH * 2);
  _Float16* v16b = (_Float16*)alloc((size_t)NN * DH * 2);
  _Float16* cat16 = (_Float16*)alloc((size_t)NN * 512 * 2);
  _Float16* fused16 = (_Float16*)alloc((size_t)NN * 512 * 2);
  _Float16* z1 = (_Float16*)alloc((size_t)NN * 128 * 2);
  _Float16* z2 = (_Float16*)alloc((size_t)NN * 64 * 2);

  const float NEG_INF = -__builtin_inff();
  auto gemm = [&](const _Float16* A, const _Float16* BT, const float* bias,
                  float* o32, _Float16* o16, int M, int Nc, int K, int act) {
    dim3 g((M + 127) / 128, Nc / 64);
    k_gemm_wmma<<<g, 256, 0, stream>>>(A, BT, bias, o32, o16, M, Nc, K, act);
  };
  auto tr = [&](const float* s, int ld, int off, int K, int Nc, _Float16* d) {
    int tot = K * Nc;
    k_transpose16<<<(tot + 255) / 256, 256, 0, stream>>>(s, ld, off, K, Nc, d);
  };
  auto fill = [&](float* p, float v, int n) {
    k_fill_f32<<<(n + 255) / 256, 256, 0, stream>>>(p, v, n);
  };
  auto ln = [&](const float* a, const float* b, const float* bias, const float* g,
                const float* be, const float* resid, float* o32, int l32,
                _Float16* o16, int l16) {
    k_ln<<<(NN * 32 + 255) / 256, 256, 0, stream>>>(a, b, bias, g, be, resid,
                                                    o32, l32, o16, l16, NN, DH);
  };

  // ---- P0: precision staging ----
  k_f32_to_f16<<<((NN * FEAT) + 255) / 256, 256, 0, stream>>>(x, x16, NN * FEAT);
  tr(ns_in_w, DH, 0, FEAT, DH, wt_nsin);   tr(ns0_w1, HC4, 0, DH, HC4, wt_n0w1);
  tr(ns0_w2, DH, 0, HC4, DH, wt_n0w2);     tr(ns1_w1, HC4, 0, DH, HC4, wt_n1w1);
  tr(ns1_w2, DH, 0, HC4, DH, wt_n1w2);     tr(ss_in_w, DH, 0, FEAT, DH, wt_ssin);
  tr(g1_w, HC4, 0, DH, HC4, wt_g1);        tr(g2_w, DH, 0, HC4, DH, wt_g2);
  tr(m1_wqkv, 3 * DH, 2 * DH, DH, DH, wt_m1v);  tr(m1_wo, DH, 0, DH, DH, wt_m1o);
  tr(m2_wqkv, 3 * DH, 2 * DH, DH, DH, wt_m2v);  tr(m2_wo, DH, 0, DH, DH, wt_m2o);
  tr(fu_w, 512, 0, 512, 512, wt_fu);       tr(cl_w1, 128, 0, 512, 128, wt_cl1);
  tr(cl_w2, 64, 0, 128, 64, wt_cl2);

  // ---- node feature stream ----
  gemm(x16, wt_nsin, ns_in_b, h32, h16, NN, DH, FEAT, 0);                 // h
  gemm(h16, wt_n0w1, ns0_b1, nullptr, big16, NN, HC4, DH, 1);             // gelu(h@w1)
  gemm(big16, wt_n0w2, ns0_b2, tmp32, nullptr, NN, DH, HC4, 0);           // br
  ln(tmp32, nullptr, nullptr, ns0_lng, ns0_lnb, h32, h32, DH, h16, DH);   // h += LN(br)
  gemm(h16, wt_n1w1, ns1_b1, nullptr, big16, NN, HC4, DH, 1);
  gemm(big16, wt_n1w2, ns1_b2, tmp32, nullptr, NN, DH, HC4, 0);
  ln(tmp32, nullptr, nullptr, ns1_lng, ns1_lnb, h32, h32, DH, h16, DH);
  ln(h32, nullptr, nullptr, ns_out_g, ns_out_b, nullptr, node32, DH, node16, DH); // node

  // ---- structural stream: GAT layer 1 (H=4, C=256) ----
  gemm(x16, wt_ssin, ss_in_b, nullptr, h16, NN, DH, FEAT, 0);             // s (reuses h16)
  gemm(h16, wt_g1, nullptr, nullptr, big16, NN, HC4, DH, 0);              // xw1 (f16)
  k_att_reduce<<<(NN * NH * 32 + 255) / 256, 256, 0, stream>>>(big16, g1_as, g1_ad,
                                                               a_s1, a_d1, NN, NH, DH);
  fill(mx1, NEG_INF, NN * NH); fill(dn1, 0.0f, NN * NH); fill(big32, 0.0f, NN * HC4);
  k_edge_max<<<(ET * NH + 255) / 256, 256, 0, stream>>>(ei, E0, ET, a_s1, a_d1, mx1, NH);
  k_edge_expsum<<<(ET * NH + 255) / 256, 256, 0, stream>>>(ei, E0, ET, a_s1, a_d1, mx1, dn1, ee1, NH);
  k_edge_scatter<<<ET, 256, 0, stream>>>(ei, E0, big16, ee1, dn1, big32, NH, DH);
  k_bias_elu_f16<<<((NN * HC4) + 255) / 256, 256, 0, stream>>>(big32, g1_bias, big16, NN, HC4); // s2=elu(out+b)

  // ---- GAT layer 2 (H=1, C=256) ----
  gemm(big16, wt_g2, nullptr, nullptr, xw2_16, NN, DH, HC4, 0);           // xw2
  k_att_reduce<<<(NN * 32 + 255) / 256, 256, 0, stream>>>(xw2_16, g2_as, g2_ad, a_s2, a_d2, NN, 1, DH);
  fill(mx2, NEG_INF, NN); fill(dn2, 0.0f, NN); fill(big32, 0.0f, NN * DH);
  k_edge_max<<<(ET + 255) / 256, 256, 0, stream>>>(ei, E0, ET, a_s2, a_d2, mx2, 1);
  k_edge_expsum<<<(ET + 255) / 256, 256, 0, stream>>>(ei, E0, ET, a_s2, a_d2, mx2, dn2, ee2, 1);
  k_edge_scatter<<<ET, 256, 0, stream>>>(ei, E0, xw2_16, ee2, dn2, big32, 1, DH);
  ln(big32, nullptr, g2_bias, ss_out_g, ss_out_b, nullptr, st32, DH, st16, DH);   // struct

  // ---- cross-attention fusion (seq_len==1 => out_proj(v_proj(kv))) ----
  gemm(st16, wt_m1v, m1_bqkv + 2 * DH, nullptr, v16b, NN, DH, DH, 0);
  gemm(v16b, wt_m1o, m1_bo, tmp32, nullptr, NN, DH, DH, 0);
  ln(node32, tmp32, nullptr, fn_g, fn_b, nullptr, nullptr, DH, cat16, 512);       // node_enh
  gemm(node16, wt_m2v, m2_bqkv + 2 * DH, nullptr, v16b, NN, DH, DH, 0);
  gemm(v16b, wt_m2o, m2_bo, tmp32, nullptr, NN, DH, DH, 0);
  ln(st32, tmp32, nullptr, fs_g, fs_b, nullptr, nullptr, DH, cat16 + DH, 512);    // struct_enh
  gemm(cat16, wt_fu, fu_b, nullptr, fused16, NN, 512, 512, 0);                    // fused

  // ---- classifier ----
  gemm(fused16, wt_cl1, cl_b1, nullptr, z1, NN, 128, 512, 1);
  gemm(z1, wt_cl2, cl_b2, nullptr, z2, NN, 64, 128, 1);
  k_final_cls<<<((NN * NCLS) + 255) / 256, 256, 0, stream>>>(z2, cl_w3, cl_b3, (float*)d_out, NN);
}
